// MoELayer_91250875171366
// MI455X (gfx1250) — matrix operand order repaired and checked by reading, not compiled, not run on hardware
//
#include <hip/hip_runtime.h>

#define NUM_EXPERTS 8
#define TOP_K 2
#define B_ 4
#define T_ 2048
#define H_ 1024
#define F_ 4096
#define NTOK (B_*T_)        /* 8192 */
#define NSLOT (NTOK*TOP_K)  /* 16384 */

typedef __bf16 bf16_t;
typedef bf16_t v16bf __attribute__((ext_vector_type(16)));
typedef float  v8f   __attribute__((ext_vector_type(8)));
typedef float  f4    __attribute__((ext_vector_type(4)));

#define KT1  (H_/32)   /* 32  k-steps for stage1 */
#define KT2  (F_/32)   /* 128 k-steps for stage2 */
#define NT1  (F_/16)   /* 256 n-tiles stage1 */
#define NT2  (H_/16)   /* 64  n-tiles stage2 */
#define KT2H (KT2/2)   /* 64  k-steps per F-half */
#define NT1H (NT1/2)   /* 128 n-tiles per F-half */

static __device__ inline v8f vzero8() {
    v8f v;
    #pragma unroll
    for (int i = 0; i < 8; ++i) v[i] = 0.0f;
    return v;
}

union AB32 { v16bf v; f4 q[2]; };

// ---------------- init: zero routing counters -------------------------------
__global__ void k_init(int* ints) {
    int t = threadIdx.x;
    if (t < 32) ints[t] = 0;   // counts[8] + offs[9] + cursor[8] (padded)
}

// ---------------- router: logits, top-2, softmax, counts --------------------
__global__ void k_router(const float* __restrict__ x, const float* __restrict__ gw,
                         float* __restrict__ logits, int* __restrict__ ch_e,
                         float* __restrict__ ch_w, int* __restrict__ counts) {
    int t = blockIdx.x * blockDim.x + threadIdx.x;
    if (t >= NTOK) return;
    float acc[NUM_EXPERTS];
    #pragma unroll
    for (int e = 0; e < NUM_EXPERTS; ++e) acc[e] = 0.0f;
    const float* xr = x + (size_t)t * H_;
    for (int k = 0; k < H_; ++k) {
        float xv = xr[k];
        const float* g = gw + (size_t)k * NUM_EXPERTS;
        #pragma unroll
        for (int e = 0; e < NUM_EXPERTS; ++e) acc[e] += xv * g[e];
    }
    #pragma unroll
    for (int e = 0; e < NUM_EXPERTS; ++e) logits[(size_t)t * NUM_EXPERTS + e] = acc[e];
    // top-1 (lowest index wins ties, matching lax.top_k)
    int e0 = 0; float l0 = acc[0];
    #pragma unroll
    for (int e = 1; e < NUM_EXPERTS; ++e) if (acc[e] > l0) { l0 = acc[e]; e0 = e; }
    int e1 = (e0 == 0) ? 1 : 0; float l1 = acc[e1];
    #pragma unroll
    for (int e = 0; e < NUM_EXPERTS; ++e)
        if (e != e0 && acc[e] > l1) { l1 = acc[e]; e1 = e; }
    float p1 = __expf(l1 - l0);
    float inv = 1.0f / (1.0f + p1);
    ch_e[2 * t] = e0;  ch_e[2 * t + 1] = e1;
    ch_w[2 * t] = inv; ch_w[2 * t + 1] = p1 * inv;
    atomicAdd(&counts[e0], 1);
    atomicAdd(&counts[e1], 1);
}

// ---------------- exclusive scan over 8 expert counts -----------------------
__global__ void k_scan(const int* __restrict__ counts, int* __restrict__ offs) {
    if (threadIdx.x == 0 && blockIdx.x == 0) {
        int s = 0;
        for (int e = 0; e < NUM_EXPERTS; ++e) { offs[e] = s; s += counts[e]; }
        offs[NUM_EXPERTS] = s;
    }
}

// ---------------- build compact per-expert token lists ----------------------
__global__ void k_assign(const int* __restrict__ ch_e, const int* __restrict__ offs,
                         int* __restrict__ cursor, int* __restrict__ tok_list,
                         int* __restrict__ slot_of) {
    int t = blockIdx.x * blockDim.x + threadIdx.x;
    if (t >= NTOK) return;
    #pragma unroll
    for (int kk = 0; kk < TOP_K; ++kk) {
        int e = ch_e[2 * t + kk];
        int pos = atomicAdd(&cursor[e], 1);
        int slot = offs[e] + pos;
        tok_list[slot] = t;
        slot_of[2 * t + kk] = slot;
    }
}

// ---------------- x -> bf16 --------------------------------------------------
__global__ void k_cvt_x(const float* __restrict__ src, bf16_t* __restrict__ dst, size_t n) {
    for (size_t i = (size_t)blockIdx.x * blockDim.x + threadIdx.x; i < n;
         i += (size_t)gridDim.x * blockDim.x)
        dst[i] = (bf16_t)src[i];
}

// ------- weights -> bf16, swizzled into WMMA B-fragment layout --------------
// dst block (e, nt, kt) = 32 lanes x 16 bf16 contiguous.
// lane l<16:  n = nt*16+l,     k = kt*32 + i (i<8), kt*32 + 16 + i-8 (i>=8)
// lane l>=16: n = nt*16+l-16,  k = kt*32 + 8 + i (i<8), kt*32 + 24 + i-8 (i>=8)
template <int KD, int ND>
__global__ void k_cvt_w(const float* __restrict__ src, bf16_t* __restrict__ dst) {
    constexpr int KTn = KD >> 5;
    constexpr int NTn = ND >> 4;
    constexpr size_t n = (size_t)NUM_EXPERTS * KD * ND;
    for (size_t q = (size_t)blockIdx.x * blockDim.x + threadIdx.x; q < n;
         q += (size_t)gridDim.x * blockDim.x) {
        unsigned blockId = (unsigned)(q >> 9);
        int within = (int)(q & 511);
        int l = within >> 4, i = within & 15;
        int kt = (int)(blockId % KTn);                 // power-of-two -> mask
        int nt = (int)((blockId / KTn) % NTn);
        int e  = (int)(blockId / (KTn * NTn));
        int k  = kt * 32 + i + ((i >> 3) << 3) + ((l >> 4) << 3);        // +8 for upper lanes, +8 for i>=8
        int nn = nt * 16 + (l & 15);
        dst[q] = (bf16_t)src[((size_t)e * KD + k) * ND + nn];
    }
}

// ---------------- expert FFN: gather + WMMA GEMM x2 + ReLU ------------------
__global__ __launch_bounds__(256) void k_ffn(const bf16_t* __restrict__ xb,
                                             const bf16_t* __restrict__ w1f,
                                             const bf16_t* __restrict__ w2f,
                                             const int* __restrict__ tok_list,
                                             const int* __restrict__ counts,
                                             const int* __restrict__ offs,
                                             float* __restrict__ y) {
    // LDS: X tile and h tile stored in WMMA A-fragment layout
    __shared__ __align__(16) bf16_t xg[KT1][32][16];   // 32 KB
    __shared__ __align__(16) bf16_t hf[KT2H][32][16];  // 64 KB (one F-half)
    __shared__ int s_tok[16];

    int e = blockIdx.y;
    int tile = blockIdx.x;
    int cnt = counts[e];
    if (tile * 16 >= cnt) return;
    int base = offs[e] + tile * 16;

    int tid = threadIdx.x;
    int lane = tid & 31, wv = tid >> 5;

    if (tid < 16) {
        int rel = tile * 16 + tid;
        s_tok[tid] = (rel < cnt) ? tok_list[base + tid] : -1;
    }
    __syncthreads();

    // gather X rows into A-fragment layout (branchless index math):
    // m = l & 15 ; k = kt*32 + i + 8*((i>>3)&1) + 8*((l>>4)&1)
    for (int q = tid; q < KT1 * 512; q += 256) {
        int kt = q >> 9;
        int r  = q & 511;
        int l  = r >> 4, i = r & 15;
        int m  = l & 15;
        int k  = kt * 32 + i + (((i >> 3) & 1) << 3) + (((l >> 4) & 1) << 3);
        int tok = s_tok[m];
        bf16_t v = (bf16_t)0.0f;
        if (tok >= 0) v = xb[(size_t)tok * H_ + k];
        xg[kt][l][i] = v;
    }

    v8f acc2[8];
    #pragma unroll
    for (int j = 0; j < 8; ++j) acc2[j] = vzero8();

    int ncol = lane & 15;
    int mh   = (lane >> 4) << 3;

    for (int fh = 0; fh < 2; ++fh) {
        __syncthreads();   // gather done (fh=0) / stage-2 reads done (fh=1)

        // ---- stage 1: h = relu(X @ W1) for this F-half; 16 n-tiles per wave
        for (int j = 0; j < 16; ++j) {
            int nth = wv * 16 + j;       // tile within half: 0..127
            int nt  = fh * NT1H + nth;   // global F tile
            v8f acc = vzero8();
            const bf16_t* wb = w1f + ((size_t)(e * NT1 + nt) * KT1) * 512 + (size_t)lane * 16;
            #pragma unroll 8
            for (int kt = 0; kt < KT1; ++kt) {
                __builtin_prefetch(wb + (size_t)(kt + 8) * 512, 0, 1);
                AB32 A, Bv;
                const f4* ap = (const f4*)(&xg[kt][lane][0]);
                A.q[0] = ap[0]; A.q[1] = ap[1];
                const f4* bp = (const f4*)(wb + (size_t)kt * 512);
                Bv.q[0] = bp[0]; Bv.q[1] = bp[1];
                acc = __builtin_amdgcn_wmma_f32_16x16x32_bf16(
                    false, A.v, false, Bv.v, (short)0, acc, false, false);
            }
            // ReLU, convert, scatter into h A-fragment layout (branchless):
            // l2off = 16*bit3(kp) ; i2 = kp - 8*(bit3(kp)+bit4(kp))
            int c   = nth * 16 + ncol;   // column within this F-half: 0..2047
            int kth = c >> 5;            // h k-step within half
            int kp  = c & 31;
            int b3  = (kp >> 3) & 1;
            int b4  = (kp >> 4) & 1;
            int l2off = b3 << 4;
            int i2    = kp - ((b3 + b4) << 3);
            #pragma unroll
            for (int r = 0; r < 8; ++r) {
                float v = acc[r];
                v = (v > 0.0f) ? v : 0.0f;
                hf[kth][(r + mh) + l2off][i2] = (bf16_t)v;
            }
        }
        __syncthreads();

        // ---- stage 2: acc2 += h @ W2 over this F-half; 8 n-tiles per wave
        for (int j = 0; j < 8; ++j) {
            int ot = wv * 8 + j;         // H n-tile: 0..63
            const bf16_t* wb = w2f + ((size_t)(e * NT2 + ot) * KT2 + fh * KT2H) * 512
                               + (size_t)lane * 16;
            #pragma unroll 8
            for (int kt = 0; kt < KT2H; ++kt) {
                __builtin_prefetch(wb + (size_t)(kt + 8) * 512, 0, 1);
                AB32 A, Bv;
                const f4* ap = (const f4*)(&hf[kt][lane][0]);
                A.q[0] = ap[0]; A.q[1] = ap[1];
                const f4* bp = (const f4*)(wb + (size_t)kt * 512);
                Bv.q[0] = bp[0]; Bv.q[1] = bp[1];
                acc2[j] = __builtin_amdgcn_wmma_f32_16x16x32_bf16(
                    false, A.v, false, Bv.v, (short)0, acc2[j], false, false);
            }
        }
    }

    // write y rows for valid tokens (fp32, one row per routed slot)
    for (int j = 0; j < 8; ++j) {
        int ot = wv * 8 + j;
        #pragma unroll
        for (int r = 0; r < 8; ++r) {
            int m = r + mh;
            if (tile * 16 + m < cnt)
                y[(size_t)(base + m) * H_ + ot * 16 + ncol] = acc2[j][r];
        }
    }
}

// ---------------- combine: out = w0*y[slot0] + w1*y[slot1] ------------------
__global__ void k_combine(const float* __restrict__ y, const int* __restrict__ slot_of,
                          const float* __restrict__ ch_w, float* __restrict__ out) {
    const size_t n = (size_t)NTOK * H_;
    for (size_t i = (size_t)blockIdx.x * blockDim.x + threadIdx.x; i < n;
         i += (size_t)gridDim.x * blockDim.x) {
        size_t t = i >> 10;        // H_ == 1024
        size_t h = i & 1023;
        int s0 = slot_of[2 * t], s1 = slot_of[2 * t + 1];
        out[i] = ch_w[2 * t] * y[((size_t)s0 << 10) + h] +
                 ch_w[2 * t + 1] * y[((size_t)s1 << 10) + h];
    }
}

extern "C" void kernel_launch(void* const* d_in, const int* in_sizes, int n_in,
                              void* d_out, int out_size, void* d_ws, size_t ws_size,
                              hipStream_t stream) {
    const float* x  = (const float*)d_in[0];
    const float* gw = (const float*)d_in[1];
    const float* w1 = (const float*)d_in[2];
    const float* w2 = (const float*)d_in[3];
    float* out = (float*)d_out;
    float* logits = out + (size_t)NTOK * H_;

    // workspace carve-up
    char* w = (char*)d_ws;
    size_t o = 0;
    auto alloc = [&](size_t bytes) -> char* {
        char* p = w + o;
        o += (bytes + 255) & ~(size_t)255;
        return p;
    };
    bf16_t* xb   = (bf16_t*)alloc((size_t)NTOK * H_ * sizeof(bf16_t));
    bf16_t* w1f  = (bf16_t*)alloc((size_t)NUM_EXPERTS * H_ * F_ * sizeof(bf16_t));
    bf16_t* w2f  = (bf16_t*)alloc((size_t)NUM_EXPERTS * F_ * H_ * sizeof(bf16_t));
    float*  y    = (float*)alloc((size_t)NSLOT * H_ * sizeof(float));
    int*    ints = (int*)alloc(64 * sizeof(int));
    int* counts = ints;          // [8]
    int* offs   = ints + 8;      // [9]
    int* cursor = ints + 17;     // [8]
    int*   ch_e     = (int*)alloc((size_t)NSLOT * sizeof(int));
    float* ch_w     = (float*)alloc((size_t)NSLOT * sizeof(float));
    int*   tok_list = (int*)alloc((size_t)NSLOT * sizeof(int));
    int*   slot_of  = (int*)alloc((size_t)NSLOT * sizeof(int));

    k_init<<<1, 32, 0, stream>>>(ints);
    k_router<<<(NTOK + 127) / 128, 128, 0, stream>>>(x, gw, logits, ch_e, ch_w, counts);
    k_scan<<<1, 1, 0, stream>>>(counts, offs);
    k_assign<<<(NTOK + 127) / 128, 128, 0, stream>>>(ch_e, offs, cursor, tok_list, slot_of);

    k_cvt_x<<<8192, 256, 0, stream>>>(x, xb, (size_t)NTOK * H_);
    k_cvt_w<H_, F_><<<16384, 256, 0, stream>>>(w1, w1f);
    k_cvt_w<F_, H_><<<16384, 256, 0, stream>>>(w2, w2f);

    k_ffn<<<dim3(512, NUM_EXPERTS), 256, 0, stream>>>(xb, w1f, w2f, tok_list, counts, offs, y);

    k_combine<<<8192, 256, 0, stream>>>(y, slot_of, ch_w, out);
}
